// MultiHeadSelfAttention_7670811590877
// MI455X (gfx1250) — matrix operand order repaired and checked by reading, not compiled, not run on hardware
//
#include <hip/hip_runtime.h>
#include <cstdint>

// ---------------- problem constants ----------------
#define L_SEQ   4096
#define NBATCH  2
#define HIDDEN  1024
#define DH      64
#define NB      64          // number of 64-wide key blocks
#define ZDIM    32          // NBATCH * HEADS
#define MROWS   8192        // L_SEQ * NBATCH

typedef __attribute__((ext_vector_type(16))) __bf16 v16bf;
typedef __attribute__((ext_vector_type(8)))  float  v8f;
typedef __attribute__((ext_vector_type(4)))  unsigned int v4u;
typedef __attribute__((ext_vector_type(8)))  int v8i;
typedef __attribute__((ext_vector_type(4)))  int v4i;
typedef __attribute__((address_space(3))) char lds_char;

union ABFrag { v16bf v; uint4 q[2]; };

__device__ __forceinline__ __bf16 f2bf(float f) {
  unsigned u = __builtin_bit_cast(unsigned, f);
  unsigned r = (u + 0x7fffu + ((u >> 16) & 1u)) >> 16;
  unsigned short s = (unsigned short)r;
  return __builtin_bit_cast(__bf16, s);
}

__device__ __forceinline__ uint4 ld16(const void* p) { return *(const uint4*)p; }

__device__ __forceinline__ v8f wmma_bf16(const ABFrag& a, const ABFrag& b, v8f c) {
  return __builtin_amdgcn_wmma_f32_16x16x32_bf16(false, a.v, false, b.v,
                                                 (short)0, c, false, false);
}

// LDS byte offset (within this workgroup's allocation) of a shared-memory pointer
__device__ __forceinline__ unsigned lds_off(void* p) {
  return (unsigned)(unsigned long long)(lds_char*)p;
}

// ---- TDM: DMA a 2-D bf16 tile (tile_d1 rows x tile_d0 elems) into LDS, with
// row padding of 8 DWORDs after every 32 DWORDs (128B row -> 160B LDS stride).
__device__ __forceinline__ void tdm_load_tile(
    unsigned lds_byte_off, const void* gptr,
    unsigned tensor_d0, unsigned tensor_d1,
    unsigned tile_d0, unsigned tile_d1, unsigned stride0_elems)
{
  unsigned long long ga = (unsigned long long)(size_t)gptr;
  v4u g0;
  g0[0] = 1u;                                   // count=1 (valid descriptor)
  g0[1] = lds_byte_off;                         // lds_addr
  g0[2] = (unsigned)(ga & 0xffffffffu);         // global_addr[31:0]
  g0[3] = (unsigned)((ga >> 32) & 0x01ffffffu)  // global_addr[56:32]
          | (2u << 30);                         // type=2 ("image")
  v8i g1;
  g1[0] = (int)((1u << 16)    // data_size = 1 -> 2 bytes
              | (1u << 20)    // pad_enable
              | (4u << 22)    // pad_interval: 32 DWORDs (=128B = one row)
              | (7u << 25));  // pad_amount: 8 DWORDs (=32B)
  g1[1] = (int)((tensor_d0 & 0xffffu) << 16);                         // td0 lo16
  g1[2] = (int)((tensor_d0 >> 16) | ((tensor_d1 & 0xffffu) << 16));   // td0 hi / td1 lo
  g1[3] = (int)((tensor_d1 >> 16) | (tile_d0 << 16));                 // td1 hi / tile0
  g1[4] = (int)(tile_d1 & 0xffffu);                                   // tile1 (tile2=0)
  g1[5] = (int)stride0_elems;                                         // dim0 stride lo32
  g1[6] = 0;                                                          // stride hi / s1 lo
  g1[7] = 0;
  v4i g2 = {0, 0, 0, 0};
  v4i g3 = {0, 0, 0, 0};
  v8i g4 = {0, 0, 0, 0, 0, 0, 0, 0};
  __builtin_amdgcn_tensor_load_to_lds(g0, g1, g2, g3, g4, 0);
}

// ---------------- kernel 0: fp32 -> bf16 conversion (scale folded into Wq/bq) ----
__global__ __launch_bounds__(256) void convert_kernel(
    const float* __restrict__ x,
    const float* __restrict__ wq, const float* __restrict__ bq,
    const float* __restrict__ wk, const float* __restrict__ bk,
    const float* __restrict__ wv, const float* __restrict__ bv,
    const float* __restrict__ wo, const float* __restrict__ bo,
    __bf16* __restrict__ xb, __bf16* __restrict__ wqkv, float* __restrict__ bqkv,
    __bf16* __restrict__ wob, float* __restrict__ bob)
{
  const float S = 0.125f;  // dh^-0.5 = 64^-0.5
  size_t tid = (size_t)blockIdx.x * blockDim.x + threadIdx.x;
  size_t stride = (size_t)gridDim.x * blockDim.x;
  for (size_t i = tid; i < (size_t)MROWS * HIDDEN; i += stride)
    xb[i] = f2bf(x[i]);
  const size_t W = (size_t)HIDDEN * HIDDEN;
  for (size_t i = tid; i < W; i += stride) {
    wqkv[i]         = f2bf(wq[i] * S);
    wqkv[W + i]     = f2bf(wk[i]);
    wqkv[2 * W + i] = f2bf(wv[i]);
    wob[i]          = f2bf(wo[i]);
  }
  for (size_t i = tid; i < HIDDEN; i += stride) {
    bqkv[i]        = bq[i] * S;
    bqkv[1024 + i] = bk[i];
    bqkv[2048 + i] = bv[i];
    bob[i]         = bo[i];
  }
}

// ---------------- GEMM: C[M,N] = A[M,K] @ B[N,K]^T + bias ----------------
// Block tile 128(M) x 256(N); 8 waves as 2x4, each computing a 64x64 tile.
// Tiles are DMA'd into LDS by the Tensor Data Mover, double-buffered.
// mode 0: N=3072, scatter-epilogue into q[Z][L][64], k[Z][L][64], vT[Z][64][L] (bf16)
// mode 1: N=1024, dense fp32 output
__global__ __launch_bounds__(256) void gemm_bf16_kernel(
    const __bf16* __restrict__ A, const __bf16* __restrict__ B,
    const float* __restrict__ bias, int mode,
    __bf16* __restrict__ qout, __bf16* __restrict__ kout,
    __bf16* __restrict__ vTout, float* __restrict__ fout)
{
  // per tile: A 128 rows x 64 k-elems, B 256 rows x 64 k-elems; TDM pads each
  // 128B row with 32B -> 80-elem (160B) LDS row stride; double buffered.
  __shared__ __align__(16) __bf16 As[2][128][80];   // 40 KB
  __shared__ __align__(16) __bf16 Bs[2][256][80];   // 80 KB
  const int K = HIDDEN;
  const int bm = blockIdx.x * 128, bn = blockIdx.y * 256;
  const int tid = threadIdx.x;
  const int wave = tid >> 5, lane = tid & 31;
  const int mw = (wave >> 2) * 64, nw = (wave & 3) * 64;
  const int ln = lane & 15, hi = lane >> 4;

  v8f acc[4][4] = {};

  const int NITER = K / 64;  // 16

  if (wave == 0) {  // wave-uniform: TDM ignores EXEC, so guard at branch level
    tdm_load_tile(lds_off(&As[0][0][0]), A + (size_t)bm * K, K, MROWS, 64, 128, K);
    tdm_load_tile(lds_off(&Bs[0][0][0]), B + (size_t)bn * K, K, 3072,  64, 256, K);
  }

  for (int ki = 0; ki < NITER; ++ki) {
    const int buf = ki & 1;
    __syncthreads();  // all waves done reading buf^1 before DMA overwrites it
    if (wave == 0) {
      if (ki + 1 < NITER) {
        int k0n = (ki + 1) * 64;
        tdm_load_tile(lds_off(&As[buf ^ 1][0][0]), A + (size_t)bm * K + k0n,
                      K, MROWS, 64, 128, K);
        tdm_load_tile(lds_off(&Bs[buf ^ 1][0][0]), B + (size_t)bn * K + k0n,
                      K, 3072, 64, 256, K);
        __builtin_amdgcn_s_wait_tensorcnt((unsigned short)2);  // current pair landed
      } else {
        __builtin_amdgcn_s_wait_tensorcnt((unsigned short)0);
      }
    }
    __syncthreads();  // publish DMA'd tile to all waves

#pragma unroll
    for (int ks = 0; ks < 2; ++ks) {
      ABFrag af[4];
#pragma unroll
      for (int mt = 0; mt < 4; ++mt) {
        const __bf16* p = &As[buf][mw + mt * 16 + ln][ks * 32 + hi * 8];
        af[mt].q[0] = ld16(p); af[mt].q[1] = ld16(p + 16);
      }
      ABFrag bfr[4];
#pragma unroll
      for (int nt = 0; nt < 4; ++nt) {
        const __bf16* p = &Bs[buf][nw + nt * 16 + ln][ks * 32 + hi * 8];
        bfr[nt].q[0] = ld16(p); bfr[nt].q[1] = ld16(p + 16);
      }
#pragma unroll
      for (int mt = 0; mt < 4; ++mt)
#pragma unroll
        for (int nt = 0; nt < 4; ++nt)
          acc[mt][nt] = wmma_bf16(af[mt], bfr[nt], acc[mt][nt]);
    }
  }

#pragma unroll
  for (int mt = 0; mt < 4; ++mt)
#pragma unroll
    for (int nt = 0; nt < 4; ++nt)
#pragma unroll
      for (int r = 0; r < 8; ++r) {
        int grow = bm + mw + mt * 16 + r + hi * 8;
        int gcol = bn + nw + nt * 16 + ln;
        float val = acc[mt][nt][r] + bias[gcol];
        if (mode == 0) {
          int p = gcol >> 10, hcol = gcol & 1023;
          int head = hcol >> 6, d = hcol & 63;
          int lseq = grow >> 1, nbt = grow & 1;
          int z = nbt * 16 + head;
          __bf16 bw = f2bf(val);
          if (p == 0)      qout[((size_t)z * L_SEQ + lseq) * DH + d] = bw;
          else if (p == 1) kout[((size_t)z * L_SEQ + lseq) * DH + d] = bw;
          else             vTout[((size_t)z * DH + d) * L_SEQ + lseq] = bw;
        } else {
          fout[(size_t)grow * HIDDEN + gcol] = val;
        }
      }
}

// ---------------- flash block-sparse attention ----------------
// grid (64 qblocks, 32 z); 128 threads = 4 waves; wave w owns q-rows [16w,16w+16)
__global__ __launch_bounds__(128) void attn_kernel(
    const __bf16* __restrict__ q, const __bf16* __restrict__ kk,
    const __bf16* __restrict__ vT, const int* __restrict__ rnd,
    __bf16* __restrict__ aout)
{
  __shared__ __align__(16) __bf16 plds[4][16][80];
  const int z = blockIdx.y, qb = blockIdx.x;
  const int wave = threadIdx.x >> 5, lane = threadIdx.x & 31;
  const int ln = lane & 15, hi = lane >> 4;

  // q A-fragments (rows fixed for this wave): K = dh = 64 -> 2 frags
  ABFrag aq[2];
  {
    int mrow = qb * 64 + wave * 16 + ln;
    const __bf16* qp = q + ((size_t)z * L_SEQ + mrow) * DH;
#pragma unroll
    for (int ks = 0; ks < 2; ++ks) {
      const __bf16* p = qp + ks * 32 + hi * 8;
      aq[ks].q[0] = ld16(p); aq[ks].q[1] = ld16(p + 16);
    }
  }

  float m_[8], l_[8];
  v8f O[4] = {};
#pragma unroll
  for (int r = 0; r < 8; ++r) { m_[r] = -1e30f; l_[r] = 0.f; }

  const int nslots = (qb < 2) ? NB : 7;
  for (int s = 0; s < nslots; ++s) {
    int kb, vb;
    if (qb < 2)      { kb = vb = s; }                       // head: dense over all keys
    else if (s < 2)  { kb = vb = s; }                       // global key blocks 0,1
    else if (s < 5) {                                       // sliding window
      int j = s - 2;
      if (qb < NB - 1) { kb = vb = qb - 1 + j; }
      else { kb = (j == 0) ? -1 : 61 + j; vb = (j == 2) ? -1 : 62 + j; }
    } else {                                                // random blocks
      kb = vb = rnd[(qb - 2) * 2 + (s - 5)];
    }

    // ---- scores S (16 x 64) for this key block ----
    v8f c[4];
    if (kb < 0) {
#pragma unroll
      for (int t = 0; t < 4; ++t) c[t] = (v8f){};
    } else {
#pragma unroll
      for (int t = 0; t < 4; ++t) {
        int kcol = kb * 64 + t * 16 + ln;
        const __bf16* kp = kk + ((size_t)z * L_SEQ + kcol) * DH + hi * 8;
        ABFrag b0, b1;
        b0.q[0] = ld16(kp);      b0.q[1] = ld16(kp + 16);
        b1.q[0] = ld16(kp + 32); b1.q[1] = ld16(kp + 48);
        v8f cz = {};
        cz = wmma_bf16(aq[0], b0, cz);
        cz = wmma_bf16(aq[1], b1, cz);
        c[t] = cz;
      }
    }

    // ---- online softmax update (row reductions across the 16 lanes per row) ----
#pragma unroll
    for (int r = 0; r < 8; ++r) {
      float mv = fmaxf(fmaxf(c[0][r], c[1][r]), fmaxf(c[2][r], c[3][r]));
#pragma unroll
      for (int off = 1; off < 16; off <<= 1) mv = fmaxf(mv, __shfl_xor(mv, off, 32));
      float mn = fmaxf(m_[r], mv);
      float al = __expf(m_[r] - mn);
      m_[r] = mn;
      float rs = 0.f;
#pragma unroll
      for (int t = 0; t < 4; ++t) {
        float pe = __expf(c[t][r] - mn);
        c[t][r] = pe;
        rs += pe;
      }
#pragma unroll
      for (int off = 1; off < 16; off <<= 1) rs += __shfl_xor(rs, off, 32);
      l_[r] = l_[r] * al + rs;
#pragma unroll
      for (int t = 0; t < 4; ++t) O[t][r] *= al;
    }

    // ---- stage P (bf16) through LDS (wave-private region; same-wave LDS is
    //      in-order, so a wave-local dscnt wait suffices — no block barrier) ----
#pragma unroll
    for (int t = 0; t < 4; ++t)
#pragma unroll
      for (int r = 0; r < 8; ++r)
        plds[wave][r + hi * 8][t * 16 + ln] = f2bf(c[t][r]);
    asm volatile("s_wait_dscnt 0x0" ::: "memory");

    ABFrag pa[2];
#pragma unroll
    for (int ks = 0; ks < 2; ++ks) {
      const __bf16* pp = &plds[wave][ln][ks * 32 + hi * 8];
      pa[ks].q[0] = ld16(pp); pa[ks].q[1] = ld16(pp + 16);
    }

    // ---- O += P @ V  (V read from transposed layout; contiguous B-frag loads) ----
    if (vb >= 0) {
#pragma unroll
      for (int t = 0; t < 4; ++t) {
        int d = t * 16 + ln;
        const __bf16* vp = vT + ((size_t)z * DH + d) * L_SEQ + vb * 64 + hi * 8;
        ABFrag b0, b1;
        b0.q[0] = ld16(vp);      b0.q[1] = ld16(vp + 16);
        b1.q[0] = ld16(vp + 32); b1.q[1] = ld16(vp + 48);
        O[t] = wmma_bf16(pa[0], b0, O[t]);
        O[t] = wmma_bf16(pa[1], b1, O[t]);
      }
    }
  }

  // ---- normalize + write [seq, nbatch, head*64+d] as bf16 ----
  const int nbt = z >> 4, head = z & 15;
#pragma unroll
  for (int t = 0; t < 4; ++t)
#pragma unroll
    for (int r = 0; r < 8; ++r) {
      float val = O[t][r] / l_[r];
      int seq = qb * 64 + wave * 16 + r + hi * 8;
      int d = t * 16 + ln;
      aout[((size_t)seq * NBATCH + nbt) * HIDDEN + head * DH + d] = f2bf(val);
    }
}

// ---------------- host launcher ----------------
extern "C" void kernel_launch(void* const* d_in, const int* in_sizes, int n_in,
                              void* d_out, int out_size, void* d_ws, size_t ws_size,
                              hipStream_t stream) {
  (void)in_sizes; (void)n_in; (void)out_size; (void)ws_size;
  const float* x  = (const float*)d_in[0];
  const float* wq = (const float*)d_in[1];
  const float* bq = (const float*)d_in[2];
  const float* wk = (const float*)d_in[3];
  const float* bk = (const float*)d_in[4];
  const float* wv = (const float*)d_in[5];
  const float* bv = (const float*)d_in[6];
  const float* wo = (const float*)d_in[7];
  const float* bo = (const float*)d_in[8];
  const int*   rnd = (const int*)d_in[9];
  float* out = (float*)d_out;

  char* ws = (char*)d_ws;
  size_t off = 0;
  auto carve = [&](size_t bytes) {
    char* p = ws + off;
    off += (bytes + 255) & ~(size_t)255;
    return p;
  };
  __bf16* xb    = (__bf16*)carve((size_t)MROWS * HIDDEN * 2);          // 16 MB
  __bf16* wqkv  = (__bf16*)carve((size_t)3 * HIDDEN * HIDDEN * 2);     // 6 MB
  __bf16* wob   = (__bf16*)carve((size_t)HIDDEN * HIDDEN * 2);         // 2 MB
  float*  bqkv  = (float*)carve((size_t)3 * HIDDEN * 4);
  float*  bob   = (float*)carve((size_t)HIDDEN * 4);
  __bf16* qbuf  = (__bf16*)carve((size_t)ZDIM * L_SEQ * DH * 2);       // 16 MB
  __bf16* kbuf  = (__bf16*)carve((size_t)ZDIM * L_SEQ * DH * 2);       // 16 MB
  __bf16* vTbuf = (__bf16*)carve((size_t)ZDIM * DH * L_SEQ * 2);       // 16 MB
  __bf16* abuf  = (__bf16*)carve((size_t)MROWS * HIDDEN * 2);          // 16 MB

  convert_kernel<<<dim3(2048), dim3(256), 0, stream>>>(
      x, wq, bq, wk, bk, wv, bv, wo, bo, xb, wqkv, bqkv, wob, bob);

  // QKV projection: [8192,1024] @ [3072,1024]^T  (block tile 128x256)
  gemm_bf16_kernel<<<dim3(MROWS / 128, 3072 / 256), dim3(256), 0, stream>>>(
      xb, wqkv, bqkv, 0, qbuf, kbuf, vTbuf, nullptr);

  // block-sparse flash attention
  attn_kernel<<<dim3(NB, ZDIM), dim3(128), 0, stream>>>(
      qbuf, kbuf, vTbuf, rnd, abuf);

  // output projection: [8192,1024] @ [1024,1024]^T -> fp32 d_out
  gemm_bf16_kernel<<<dim3(MROWS / 128, HIDDEN / 256), dim3(256), 0, stream>>>(
      abuf, wob, bob, 1, nullptr, nullptr, nullptr, out);
}
